// NeuralODE_13795434955298
// MI455X (gfx1250) — matrix operand order repaired and checked by reading, not compiled, not run on hardware
//
#include <hip/hip_runtime.h>

typedef __attribute__((ext_vector_type(16))) _Float16 v16h;
typedef __attribute__((ext_vector_type(8)))  float    v8f;

#define D_IN   64
#define W_HID  128
#define T_PTS  64
#define B_TOT  1024
#define TILE_M 16

// Tsit5 coefficients
#define A21 0.161f
#define A31 (-0.008480655492356989f)
#define A32 0.335480655492357f
#define A41 2.8971530571054935f
#define A42 (-6.359448489975075f)
#define A43 4.3622954328695815f
#define A51 5.325864828439257f
#define A52 (-11.748883564062828f)
#define A53 7.4955393428898365f
#define A54 (-0.09249506636175525f)
#define A61 5.86145544294642f
#define A62 (-12.92096931784711f)
#define A63 8.159367898576159f
#define A64 (-0.071584973281401f)
#define A65 (-0.028269050394068383f)
#define CB1 0.09646076681806523f
#define CB2 0.01f
#define CB3 0.4798896504144996f
#define CB4 1.379008574103742f
#define CB5 (-3.290069515436081f)
#define CB6 2.324710524099774f

// Branch-free tanh: tanh(x) = 1 - 2/(exp(2x)+1).
// v_exp_f32 + v_rcp_f32; saturates correctly at +-1 without clamping.
__device__ __forceinline__ float fast_tanh(float x) {
  const float e = __builtin_amdgcn_exp2f(x * 2.8853900817779268f); // 2*log2(e)
  const float r = __builtin_amdgcn_rcpf(e + 1.0f);
  return __builtin_fmaf(-2.0f, r, 1.0f);
}

// A-matrix fragment (16x32 f16) from row-major f16 LDS [16][ldk].
// Lane L holds row M=L&15; lanes 0-15 -> K=0..15;
// lanes 16-31 -> K=16..31.
__device__ __forceinline__ v16h load_afrag_lds(const _Float16* base, int ldk,
                                               int lane, int kstep) {
  const int m = lane & 15, half = lane >> 4;
  const _Float16* row = base + m * ldk + kstep * 32 + 16 * half;
  v16h a;
#pragma unroll
  for (int p = 0; p < 8; ++p) {
    const int k = 2 * p;
    a[2 * p]     = row[k];
    a[2 * p + 1] = row[k + 1];
  }
  return a;
}

// B-matrix fragment (32x16 f16) for column n from row-major f32 weight [N][K]
// (B[k][n] = W[n][k]).  Lane L holds column N; lanes 0-15 -> K=0..15,
// lanes 16-31 -> K=16..31, consecutive-K pairs per VGPR.
__device__ __forceinline__ v16h load_bfrag_global(const float* w, int ldk,
                                                  int n, int lane, int kstep) {
  const int half = lane >> 4;
  const float* row = w + n * ldk + kstep * 32 + 16 * half;
  v16h b;
#pragma unroll
  for (int p = 0; p < 8; ++p) {
    b[2 * p]     = (_Float16)row[2 * p];
    b[2 * p + 1] = (_Float16)row[2 * p + 1];
  }
  return b;
}

__global__ __launch_bounds__(256)
void node_tsit5_kernel(const float* __restrict__ ts,
                       const float* __restrict__ y0,
                       const float* __restrict__ W1, const float* __restrict__ b1,
                       const float* __restrict__ W2, const float* __restrict__ b2,
                       const float* __restrict__ W3, const float* __restrict__ b3,
                       float* __restrict__ out) {
  __shared__ float    sY[TILE_M * D_IN];        // current state, f32
  __shared__ float    sK[6][TILE_M * D_IN];     // RK stage derivatives, f32
  __shared__ float    sP[TILE_M * D_IN];        // layer-3 split-K partial, f32
  __shared__ _Float16 sX[TILE_M * D_IN];        // stage input, f16 (WMMA A src)
  __shared__ _Float16 sH1[TILE_M * W_HID];      // hidden h1, f16
  __shared__ _Float16 sH2[TILE_M * W_HID];      // hidden h2, f16 (double buffer)

  const int tid  = threadIdx.x;
  const int lane = tid & 31;
  const int wave = tid >> 5;                    // 8 waves
  const int blk  = blockIdx.x;                  // batch tile: rows blk*16..+15
  const int nlo  = lane & 15;
  const int half = lane >> 4;
  const int ncol = (wave << 4) + nlo;           // output column, layers 1/2
  const int n3   = ((wave & 3) << 4) + nlo;     // output column, layer 3
  const int kh3  = wave >> 2;                   // layer-3 K half (0 or 1)

  // ---- weight fragments, converted f32->f16 once, pinned in VGPRs ----
  v16h B1f[2], B2f[4], B3f[2];
#pragma unroll
  for (int s = 0; s < 2; ++s) B1f[s] = load_bfrag_global(W1, D_IN, ncol, lane, s);
#pragma unroll
  for (int s = 0; s < 4; ++s) B2f[s] = load_bfrag_global(W2, W_HID, ncol, lane, s);
#pragma unroll
  for (int s = 0; s < 2; ++s)
    B3f[s] = load_bfrag_global(W3, W_HID, n3, lane, (kh3 << 1) + s);
  const float b1v = b1[ncol];
  const float b2v = b2[ncol];
  const float b3v = (kh3 == 0) ? b3[n3] : 0.0f;   // bias only in K-half 0

  // ---- load y0 tile; out[0] = y0 ----
  const int e0 = tid * 4;
  const float* y0t = y0 + blk * (TILE_M * D_IN);
  float*       o0  = out + blk * (TILE_M * D_IN);
#pragma unroll
  for (int j = 0; j < 4; ++j) {
    const float v = y0t[e0 + j];
    sY[e0 + j] = v;
    sX[e0 + j] = (_Float16)v;
    o0[e0 + j] = v;
  }
  __syncthreads();

  float* K1 = sK[0]; float* K2 = sK[1]; float* K3 = sK[2];
  float* K4 = sK[3]; float* K5 = sK[4]; float* K6 = sK[5];

  // f(sX) -> kdst   (one MLP eval; sX is f16 [16][64] row-major in LDS)
  auto eval_f = [&](float* kdst) {
    // layer 1: [16x64] x [64x128]; bias pre-loaded into accumulator
    v8f acc;
#pragma unroll
    for (int r = 0; r < 8; ++r) acc[r] = b1v;
#pragma unroll
    for (int s = 0; s < 2; ++s) {
      v16h a = load_afrag_lds(sX, D_IN, lane, s);
      acc = __builtin_amdgcn_wmma_f32_16x16x32_f16(false, a, false, B1f[s],
                                                   (short)0, acc, false, false);
    }
#pragma unroll
    for (int r = 0; r < 8; ++r)
      sH1[(r + 8 * half) * W_HID + ncol] = (_Float16)fast_tanh(acc[r]);
    __syncthreads();

    // layer 2: [16x128] x [128x128]; reads sH1, writes sH2 (no mid barrier)
    v8f acc2;
#pragma unroll
    for (int r = 0; r < 8; ++r) acc2[r] = b2v;
#pragma unroll
    for (int s = 0; s < 4; ++s) {
      v16h a2 = load_afrag_lds(sH1, W_HID, lane, s);
      acc2 = __builtin_amdgcn_wmma_f32_16x16x32_f16(false, a2, false, B2f[s],
                                                    (short)0, acc2, false, false);
    }
#pragma unroll
    for (int r = 0; r < 8; ++r)
      sH2[(r + 8 * half) * W_HID + ncol] = (_Float16)fast_tanh(acc2[r]);
    __syncthreads();

    // layer 3: [16x128] x [128x64], K split across wave quads (all 8 waves)
    v8f acc3;
#pragma unroll
    for (int r = 0; r < 8; ++r) acc3[r] = b3v;
#pragma unroll
    for (int s = 0; s < 2; ++s) {
      v16h a3 = load_afrag_lds(sH2, W_HID, lane, (kh3 << 1) + s);
      acc3 = __builtin_amdgcn_wmma_f32_16x16x32_f16(false, a3, false, B3f[s],
                                                    (short)0, acc3, false, false);
    }
    float* dst3 = (kh3 == 0) ? kdst : sP;
#pragma unroll
    for (int r = 0; r < 8; ++r)
      dst3[(r + 8 * half) * D_IN + n3] = acc3[r];
    __syncthreads();
    // reduce split-K partials (each thread owns elements e0..e0+3 thereafter)
#pragma unroll
    for (int j = 0; j < 4; ++j) kdst[e0 + j] += sP[e0 + j];
  };

  for (int iv = 0; iv < T_PTS - 1; ++iv) {
    const float h = (ts[iv + 1] - ts[iv]) * 0.5f;   // SUBSTEPS = 2
#pragma unroll 1
    for (int sub = 0; sub < 2; ++sub) {
      eval_f(K1);
#pragma unroll
      for (int j = 0; j < 4; ++j) {
        const int e = e0 + j;
        sX[e] = (_Float16)(sY[e] + h * (A21 * K1[e]));
      }
      __syncthreads();
      eval_f(K2);
#pragma unroll
      for (int j = 0; j < 4; ++j) {
        const int e = e0 + j;
        sX[e] = (_Float16)(sY[e] + h * (A31 * K1[e] + A32 * K2[e]));
      }
      __syncthreads();
      eval_f(K3);
#pragma unroll
      for (int j = 0; j < 4; ++j) {
        const int e = e0 + j;
        sX[e] = (_Float16)(sY[e] + h * (A41 * K1[e] + A42 * K2[e] + A43 * K3[e]));
      }
      __syncthreads();
      eval_f(K4);
#pragma unroll
      for (int j = 0; j < 4; ++j) {
        const int e = e0 + j;
        sX[e] = (_Float16)(sY[e] + h * (A51 * K1[e] + A52 * K2[e] +
                                        A53 * K3[e] + A54 * K4[e]));
      }
      __syncthreads();
      eval_f(K5);
#pragma unroll
      for (int j = 0; j < 4; ++j) {
        const int e = e0 + j;
        sX[e] = (_Float16)(sY[e] + h * (A61 * K1[e] + A62 * K2[e] + A63 * K3[e] +
                                        A64 * K4[e] + A65 * K5[e]));
      }
      __syncthreads();
      eval_f(K6);
#pragma unroll
      for (int j = 0; j < 4; ++j) {
        const int e = e0 + j;
        const float v = sY[e] + h * (CB1 * K1[e] + CB2 * K2[e] + CB3 * K3[e] +
                                     CB4 * K4[e] + CB5 * K5[e] + CB6 * K6[e]);
        sY[e] = v;
        sX[e] = (_Float16)v;
      }
      __syncthreads();
    }
    float* ot = out + (size_t)(iv + 1) * (B_TOT * D_IN) + blk * (TILE_M * D_IN);
#pragma unroll
    for (int j = 0; j < 4; ++j) ot[e0 + j] = sY[e0 + j];
  }
}

extern "C" void kernel_launch(void* const* d_in, const int* in_sizes, int n_in,
                              void* d_out, int out_size, void* d_ws, size_t ws_size,
                              hipStream_t stream) {
  const float* ts = (const float*)d_in[0];
  const float* y0 = (const float*)d_in[1];
  const float* W1 = (const float*)d_in[2];
  const float* b1 = (const float*)d_in[3];
  const float* W2 = (const float*)d_in[4];
  const float* b2 = (const float*)d_in[5];
  const float* W3 = (const float*)d_in[6];
  const float* b3 = (const float*)d_in[7];
  node_tsit5_kernel<<<B_TOT / TILE_M, 256, 0, stream>>>(
      ts, y0, W1, b1, W2, b2, W3, b3, (float*)d_out);
}
